// HebbianMemory_26834955665947
// MI455X (gfx1250) — matrix operand order repaired and checked by reading, not compiled, not run on hardware
//
#include <hip/hip_runtime.h>
#include <cstddef>
#include <cstdint>

// ---------------------------------------------------------------------------
// HebbianMemory on MI455X (gfx1250, wave32, WMMA + async global->LDS)
//
//   1) k/v/q projections: bf16 WMMA GEMM (f32 accum) -> bf16 outputs.
//   2) Scan: exact chunked reformulation (T=32). Per workgroup: one batch x
//      64 state columns. f32 state lives in VGPR accumulators (C/D tiles of
//      v_wmma_f32_16x16x32_bf16); bf16 mirror in LDS feeds the Q*S_prev
//      history WMMA. Chunk staging uses GLOBAL_LOAD_ASYNC_TO_LDS (ASYNCcnt)
//      since q/k/v are already bf16; next chunk is prefetched with
//      global_prefetch while the current chunk computes.
//   3) Output projection: bf16 WMMA GEMM into d_out (f32 + bias).
// ---------------------------------------------------------------------------

#define DIMD 512
#define NB 8
#define SEQ 2048
#define CH_T 32     // chunk length (= WMMA K for bf16)
#define ECOLS 64    // state columns per workgroup
#define LRc 0.1f
#define DECAYc 0.9f

typedef __bf16 bf16;
typedef __attribute__((ext_vector_type(16))) __bf16 v16bf;
typedef __attribute__((ext_vector_type(8)))  float  v8f;

__device__ inline v8f wmma_bf16(v16bf a, v16bf b, v8f c) {
  // D = A(16x32) * B(32x16) + C, f32 accumulate
  return __builtin_amdgcn_wmma_f32_16x16x32_bf16(false, a, false, b,
                                                 (short)0, c, false, false);
}

// Async global->LDS copies (CDNA5, tracked by ASYNCcnt). dst is a per-lane
// LDS byte address (generic LDS pointer truncated to 32 bits per ISA 10.2).
__device__ inline void async_g2l_b128(uint32_t lds, const void* g) {
  asm volatile("global_load_async_to_lds_b128 %0, %1, off"
               :: "v"(lds), "v"(g) : "memory");
}
__device__ inline void async_g2l_b64(uint32_t lds, const void* g) {
  asm volatile("global_load_async_to_lds_b64 %0, %1, off"
               :: "v"(lds), "v"(g) : "memory");
}
__device__ inline void wait_async0() {
  asm volatile("s_wait_asynccnt 0x0" ::: "memory");
}

// A-matrix 16x32 bf16 fragment. base points at element (row0, k0) of a
// row-major matrix with leading dimension ld.
// Lane l: m = l&15, half = l>>4; element e holds K = 16*(e>>3) + 8*half + (e&7).
__device__ inline v16bf ld_frag_a(const bf16* base, int ld, int lane) {
  int m = lane & 15, h = lane >> 4;
  const bf16* p = base + m * ld;
  v16bf a;
#pragma unroll
  for (int e = 0; e < 16; ++e) {
    int kk = 16 * (e >> 3) + 8 * h + (e & 7);
    a[e] = p[kk];
  }
  return a;
}

// B-matrix 32x16 bf16 fragment from a row-major (K x N) source, base at (k0,n0).
// Lane l: n = l&15, half = l>>4; element e holds K = 16*(e>>3) + 8*half + (e&7).
__device__ inline v16bf ld_frag_b(const bf16* base, int ld, int lane) {
  int n = lane & 15, h = lane >> 4;
  v16bf b;
#pragma unroll
  for (int e = 0; e < 16; ++e) {
    b[e] = base[(16 * (e >> 3) + 8 * h + (e & 7)) * ld + n];
  }
  return b;
}

// B-matrix 32x16 fragment from a TRANSPOSED source: B[k][n] = base[n*ld + k].
__device__ inline v16bf ld_frag_bT(const bf16* base, int ld, int lane) {
  int n = lane & 15, h = lane >> 4;
  v16bf b;
#pragma unroll
  for (int e = 0; e < 16; ++e) {
    b[e] = base[n * ld + 16 * (e >> 3) + 8 * h + (e & 7)];
  }
  return b;
}

// ---------------------------------------------------------------------------
__global__ void cvt_f32_bf16(const float* __restrict__ in,
                             bf16* __restrict__ out, int nelem) {
  for (int i = blockIdx.x * blockDim.x + threadIdx.x; i < nelem;
       i += gridDim.x * blockDim.x)
    out[i] = (bf16)in[i];
}

// C[MxD] = A[MxD] @ W^T + bias, W row-major (out,in). 256 thr / 8 waves,
// each wave computes a 16x64 output strip; 1 block per 16 rows.
// OutT = bf16 for the k/v/q projections (feeds WMMA later), float for out.
template <typename OutT>
__global__ __launch_bounds__(256) void gemm_xwT_bias(
    const bf16* __restrict__ A, const bf16* __restrict__ W,
    const float* __restrict__ bias, OutT* __restrict__ C) {
  const int lane = threadIdx.x & 31;
  const int wv   = threadIdx.x >> 5;     // 0..7
  const int row0 = blockIdx.x * 16;
  const int n0   = wv * 64;
  v8f acc[4] = {};
  for (int kk = 0; kk < DIMD; kk += 32) {
    v16bf a = ld_frag_a(A + (size_t)row0 * DIMD + kk, DIMD, lane);
#pragma unroll
    for (int nt = 0; nt < 4; ++nt) {
      v16bf b = ld_frag_bT(W + (size_t)(n0 + nt * 16) * DIMD + kk, DIMD, lane);
      acc[nt] = wmma_bf16(a, b, acc[nt]);
    }
  }
  const int n = lane & 15, h = lane >> 4;
#pragma unroll
  for (int nt = 0; nt < 4; ++nt) {
    float bv = bias[n0 + nt * 16 + n];
#pragma unroll
    for (int r = 0; r < 8; ++r)
      C[(size_t)(row0 + r + 8 * h) * DIMD + n0 + nt * 16 + n] =
          (OutT)(acc[nt][r] + bv);
  }
}

// ---------------------------------------------------------------------------
// Chunked Hebbian scan. grid = NB * (DIMD/ECOLS) blocks of 512 threads
// (16 waves). Wave w owns state rows [32w, 32w+32) x this block's 64 columns,
// held as 2x4 f32 WMMA accumulator tiles (64 VGPRs/lane).
__global__ __launch_bounds__(512) void hebbian_scan(
    const bf16* __restrict__ qg, const bf16* __restrict__ kg,
    const bf16* __restrict__ vg, const float* __restrict__ st_in,
    bf16* __restrict__ out_bf, float* __restrict__ st_out) {

  __shared__ __align__(16) bf16  Qr[CH_T][DIMD];     // 32 KB q chunk
  __shared__ __align__(16) bf16  Kr[CH_T][DIMD];     // 32 KB k chunk
  __shared__ __align__(16) bf16  Vb[CH_T][ECOLS];    //  4 KB v slice
  __shared__ __align__(16) float Araw[CH_T][CH_T];   //  4 KB q.k dots
  __shared__ __align__(16) bf16  Amask[CH_T][CH_T];  //  2 KB masked A
  __shared__ __align__(16) float Obuf[CH_T][ECOLS];  //  8 KB output accum
  __shared__ __align__(16) bf16  Sb[DIMD][ECOLS];    // 64 KB bf16 state mirror

  const int tid  = threadIdx.x;
  const int lane = tid & 31;
  const int wv   = tid >> 5;             // 0..15
  const int b    = blockIdx.x >> 3;      // batch
  const int cb   = blockIdx.x & 7;       // column block
  const int e0   = cb * ECOLS;
  const int k0   = wv * 32;              // wave's d-row base
  const int n = lane & 15, h = lane >> 4;

  // LDS byte addresses for the async copy destinations (ISA 10.2: generic
  // LDS address truncated to 32 bits is the LDS byte offset).
  const uint32_t qlds = (uint32_t)(uintptr_t)&Qr[0][0];
  const uint32_t klds = (uint32_t)(uintptr_t)&Kr[0][0];
  const uint32_t vlds = (uint32_t)(uintptr_t)&Vb[0][0];

  // chunk-invariant decay scale factors
  float qsc[2];                              // decay^tau on q rows
  qsc[0] = __powf(DECAYc, (float)n);
  qsc[1] = __powf(DECAYc, (float)(16 + n));
  float ksc[16];                             // lr*decay^(31-tau) on k cols
#pragma unroll
  for (int e = 0; e < 16; ++e) {
    int kt = 16 * (e >> 3) + 8 * h + (e & 7);
    ksc[e] = LRc * __powf(DECAYc, (float)(31 - kt));
  }
  const float decayT = __powf(DECAYc, (float)CH_T);
  float mfac[2];                             // causal mask factors
#pragma unroll
  for (int u = 0; u < 2; ++u) {
    int idx = tid * 2 + u;
    int t = idx >> 5, j = idx & 31;
    mfac[u] = (j < t) ? LRc * __powf(DECAYc, (float)(t - 1 - j)) : 0.f;
  }

  // ---- load initial state into f32 accumulators + bf16 LDS mirror ----
  v8f S[2][4];
#pragma unroll
  for (int rt = 0; rt < 2; ++rt)
#pragma unroll
    for (int nt = 0; nt < 4; ++nt)
#pragma unroll
      for (int r = 0; r < 8; ++r) {
        int d = k0 + rt * 16 + r + 8 * h;
        int e = e0 + nt * 16 + n;
        S[rt][nt][r] = st_in[((size_t)b * DIMD + d) * DIMD + e];
      }
#pragma unroll
  for (int rt = 0; rt < 2; ++rt)
#pragma unroll
    for (int nt = 0; nt < 4; ++nt)
#pragma unroll
      for (int r = 0; r < 8; ++r)
        Sb[k0 + rt * 16 + r + 8 * h][nt * 16 + n] = (bf16)S[rt][nt][r];
  __syncthreads();

  for (int ch = 0; ch < SEQ / CH_T; ++ch) {
    const int t0 = ch * CH_T;

    // (a) async-stage chunk (bf16 bytes, no VGPR round trip) + zero buffers
    const char* qsrc = (const char*)(qg + ((size_t)b * SEQ + t0) * DIMD);
    const char* ksrc = (const char*)(kg + ((size_t)b * SEQ + t0) * DIMD);
#pragma unroll
    for (int u = 0; u < 4; ++u) {
      int off = tid * 16 + u * 8192;     // covers 32 KB with 512 threads
      async_g2l_b128(qlds + off, qsrc + off);
      async_g2l_b128(klds + off, ksrc + off);
    }
    {
      int tt = tid >> 4, eb = (tid & 15) * 8;   // 32 rows x 128 B
      const char* vsrc =
          (const char*)(vg + ((size_t)b * SEQ + t0 + tt) * DIMD + e0);
      async_g2l_b64(vlds + tt * (ECOLS * 2) + eb, vsrc + eb);
    }
#pragma unroll
    for (int u = 0; u < 2; ++u) ((float*)Araw)[tid + 512 * u] = 0.f;
#pragma unroll
    for (int u = 0; u < 4; ++u) ((float*)Obuf)[tid + 512 * u] = 0.f;

    // prefetch next chunk's q/k into cache while we compute this one
    if (ch + 1 < SEQ / CH_T) {
      __builtin_prefetch(qsrc + CH_T * DIMD * 2 + tid * 64, 0, 0);
      __builtin_prefetch(ksrc + CH_T * DIMD * 2 + tid * 64, 0, 0);
    }
    wait_async0();
    __syncthreads();

    // (b) A_raw = Q K^T over full D (each wave its 32-d slice, ds_add reduce)
#pragma unroll
    for (int mt = 0; mt < 2; ++mt) {
      v16bf a = ld_frag_a(&Qr[mt * 16][k0], DIMD, lane);
#pragma unroll
      for (int ntt = 0; ntt < 2; ++ntt) {
        v16bf bb = ld_frag_bT(&Kr[ntt * 16][k0], DIMD, lane);
        v8f c = {};
        c = wmma_bf16(a, bb, c);
#pragma unroll
        for (int r = 0; r < 8; ++r)
          atomicAdd(&Araw[mt * 16 + r + 8 * h][ntt * 16 + n], c[r]);
      }
    }
    __syncthreads();

    // (c) causal decay mask:  A[t][j] = lr*decay^(t-1-j)*A_raw  (j<t)
#pragma unroll
    for (int u = 0; u < 2; ++u) {
      int idx = tid * 2 + u;
      ((bf16*)Amask)[idx] = (bf16)(mfac[u] * ((const float*)Araw)[idx]);
    }
    __syncthreads();

    // (d) intra-chunk: out += Amask(32x32) @ V(32x64)  (8 tiles, waves 0..7)
    if (wv < 8) {
      int mt = wv >> 2, nt = wv & 3;
      v16bf a = ld_frag_a(&Amask[mt * 16][0], CH_T, lane);
      v16bf bb = ld_frag_b(&Vb[0][nt * 16], ECOLS, lane);
      v8f c = {};
      c = wmma_bf16(a, bb, c);
#pragma unroll
      for (int r = 0; r < 8; ++r)
        atomicAdd(&Obuf[mt * 16 + r + 8 * h][nt * 16 + n], c[r]);
    }

    // (e) history: out += (decay^tau * Q) @ S_prev  (partial over wave's d)
#pragma unroll
    for (int mt = 0; mt < 2; ++mt) {
      v16bf a;
      const bf16* p = &Qr[mt * 16 + n][k0];
#pragma unroll
      for (int e = 0; e < 16; ++e) {
        int kk = 16 * (e >> 3) + 8 * h + (e & 7);
        a[e] = (bf16)((float)p[kk] * qsc[mt]);
      }
#pragma unroll
      for (int nt = 0; nt < 4; ++nt) {
        v16bf bb = ld_frag_b(&Sb[k0][nt * 16], ECOLS, lane);
        v8f c = {};
        c = wmma_bf16(a, bb, c);
#pragma unroll
        for (int r = 0; r < 8; ++r)
          atomicAdd(&Obuf[mt * 16 + r + 8 * h][nt * 16 + n], c[r]);
      }
    }
    __syncthreads();

    // (f) state update: S = decay^32 * S + (lr*decay^(31-tau)*K)^T @ V
#pragma unroll
    for (int rt = 0; rt < 2; ++rt)
#pragma unroll
      for (int nt = 0; nt < 4; ++nt)
#pragma unroll
        for (int r = 0; r < 8; ++r) S[rt][nt][r] *= decayT;
#pragma unroll
    for (int rt = 0; rt < 2; ++rt) {
      v16bf a;   // A[m][tau] = K[tau][d] scaled
#pragma unroll
      for (int e = 0; e < 16; ++e) {
        int kt = 16 * (e >> 3) + 8 * h + (e & 7);
        a[e] = (bf16)((float)Kr[kt][k0 + rt * 16 + n] * ksc[e]);
      }
#pragma unroll
      for (int nt = 0; nt < 4; ++nt) {
        v16bf bb = ld_frag_b(&Vb[0][nt * 16], ECOLS, lane);
        S[rt][nt] = wmma_bf16(a, bb, S[rt][nt]);
      }
    }
    // refresh bf16 mirror for next chunk's history term
#pragma unroll
    for (int rt = 0; rt < 2; ++rt)
#pragma unroll
      for (int nt = 0; nt < 4; ++nt)
#pragma unroll
        for (int r = 0; r < 8; ++r)
          Sb[k0 + rt * 16 + r + 8 * h][nt * 16 + n] = (bf16)S[rt][nt][r];

    // (g) flush out_pre chunk slice (bf16, feeds the final WMMA GEMM)
    bf16* ob = out_bf + ((size_t)b * SEQ + t0) * DIMD + e0;
#pragma unroll
    for (int u = 0; u < 4; ++u) {
      int i = tid + u * 512;
      int tt = i >> 6, e = i & 63;
      ob[(size_t)tt * DIMD + e] = (bf16)Obuf[tt][e];
    }
    __syncthreads();
  }

  // ---- final state ----
#pragma unroll
  for (int rt = 0; rt < 2; ++rt)
#pragma unroll
    for (int nt = 0; nt < 4; ++nt)
#pragma unroll
      for (int r = 0; r < 8; ++r) {
        int d = k0 + rt * 16 + r + 8 * h;
        int e = e0 + nt * 16 + n;
        st_out[((size_t)b * DIMD + d) * DIMD + e] = S[rt][nt][r];
      }
}

// ---------------------------------------------------------------------------
extern "C" void kernel_launch(void* const* d_in, const int* in_sizes, int n_in,
                              void* d_out, int out_size, void* d_ws,
                              size_t ws_size, hipStream_t stream) {
  const float* x  = (const float*)d_in[0];
  const float* st = (const float*)d_in[1];
  const float* Wk = (const float*)d_in[2];
  const float* bk = (const float*)d_in[3];
  const float* Wv = (const float*)d_in[4];
  const float* bv = (const float*)d_in[5];
  const float* Wq = (const float*)d_in[6];
  const float* bq = (const float*)d_in[7];
  const float* Wo = (const float*)d_in[8];
  const float* bo = (const float*)d_in[9];

  float* out    = (float*)d_out;
  float* fstate = out + (size_t)NB * SEQ * DIMD;

  char* w = (char*)d_ws;
  bf16* xbf  = (bf16*)(w + 0);           // 16 MB
  bf16* wkbf = (bf16*)(w + 16777216);    // 512 KB each
  bf16* wvbf = (bf16*)(w + 17301504);
  bf16* wqbf = (bf16*)(w + 17825792);
  bf16* wobf = (bf16*)(w + 18350080);
  bf16* kbf  = (bf16*)(w + 18874368);    // 16 MB each
  bf16* vbf  = (bf16*)(w + 35651584);
  bf16* qbf  = (bf16*)(w + 52428800);
  bf16* opbf = (bf16*)(w + 69206016);    // 16 MB

  const int NX = NB * SEQ * DIMD;  // 8388608
  const int NW = DIMD * DIMD;      // 262144

  cvt_f32_bf16<<<4096, 256, 0, stream>>>(x, xbf, NX);
  cvt_f32_bf16<<<256, 256, 0, stream>>>(Wk, wkbf, NW);
  cvt_f32_bf16<<<256, 256, 0, stream>>>(Wv, wvbf, NW);
  cvt_f32_bf16<<<256, 256, 0, stream>>>(Wq, wqbf, NW);
  cvt_f32_bf16<<<256, 256, 0, stream>>>(Wo, wobf, NW);

  gemm_xwT_bias<bf16><<<NB * SEQ / 16, 256, 0, stream>>>(xbf, wkbf, bk, kbf);
  gemm_xwT_bias<bf16><<<NB * SEQ / 16, 256, 0, stream>>>(xbf, wvbf, bv, vbf);
  gemm_xwT_bias<bf16><<<NB * SEQ / 16, 256, 0, stream>>>(xbf, wqbf, bq, qbf);

  hebbian_scan<<<NB * (DIMD / ECOLS), 512, 0, stream>>>(qbf, kbf, vbf, st,
                                                        opbf, fstate);

  gemm_xwT_bias<float><<<NB * SEQ / 16, 256, 0, stream>>>(opbf, wobf, bo, out);
}
